// SVXSoftmax_64244120813609
// MI455X (gfx1250) — matrix operand order repaired and checked by reading, not compiled, not run on hardware
//
#include <hip/hip_runtime.h>

typedef __attribute__((ext_vector_type(16))) _Float16 v16h;
typedef __attribute__((ext_vector_type(8)))  _Float16 v8h;
typedef __attribute__((ext_vector_type(8)))  float    v8f;

constexpr int   D_DIM  = 512;      // feature dim (K)
constexpr int   B_ROWS = 512;      // batch (M)
constexpr int   C_COLS = 100000;   // classes (N)
constexpr int   TILE_N = 32;       // weight rows per workgroup
constexpr float SCALE_S = 32.0f;
constexpr float MARG    = 0.35f;
constexpr float TPAR    = 0.2f;
constexpr float EPS_N   = 1e-12f;

// ---------------------------------------------------------------------------
// Kernel 1: L2-normalize input rows, emit f16 (values in [-1,1], f16 is safe).
// ---------------------------------------------------------------------------
__global__ __launch_bounds__(256) void k_norm_input(const float* __restrict__ in,
                                                    _Float16* __restrict__ a16) {
  const int row = blockIdx.x;
  const int t   = threadIdx.x;
  const float* x = in + (size_t)row * D_DIM;
  float x0 = x[t], x1 = x[t + 256];
  float s = x0 * x0 + x1 * x1;
#pragma unroll
  for (int off = 16; off > 0; off >>= 1) s += __shfl_xor(s, off, 32);
  __shared__ float red[8];
  const int wave = t >> 5, lane = t & 31;
  if (lane == 0) red[wave] = s;
  __syncthreads();
  if (t == 0) {
    float tot = 0.f;
#pragma unroll
    for (int i = 0; i < 8; ++i) tot += red[i];
    red[0] = 1.0f / fmaxf(sqrtf(tot), EPS_N);
  }
  __syncthreads();
  const float sc = red[0];
  a16[(size_t)row * D_DIM + t]       = (_Float16)(x0 * sc);
  a16[(size_t)row * D_DIM + t + 256] = (_Float16)(x1 * sc);
}

// ---------------------------------------------------------------------------
// Kernel 2: gt[b] = clip( input_hat[b] . weight[label[b]] / ||weight[label[b]]|| )
// 512 tiny dot products; avoids any cross-workgroup dependency in the GEMM.
// ---------------------------------------------------------------------------
__global__ __launch_bounds__(256) void k_gt(const _Float16* __restrict__ a16,
                                            const float* __restrict__ weight,
                                            const long long* __restrict__ label,
                                            float* __restrict__ gt) {
  const int row = blockIdx.x;
  const int t   = threadIdx.x;
  const long long lab = label[row];
  const float*    w   = weight + (size_t)lab * D_DIM;
  const _Float16* a   = a16 + (size_t)row * D_DIM;
  float dot = 0.f, wsq = 0.f;
  for (int k = t; k < D_DIM; k += 256) {
    float wv = w[k];
    float av = (float)a[k];
    dot += av * wv;
    wsq += wv * wv;
  }
#pragma unroll
  for (int off = 16; off > 0; off >>= 1) {
    dot += __shfl_xor(dot, off, 32);
    wsq += __shfl_xor(wsq, off, 32);
  }
  __shared__ float rd[8], rw[8];
  const int wave = t >> 5, lane = t & 31;
  if (lane == 0) { rd[wave] = dot; rw[wave] = wsq; }
  __syncthreads();
  if (t == 0) {
    float td = 0.f, tw = 0.f;
#pragma unroll
    for (int i = 0; i < 8; ++i) { td += rd[i]; tw += rw[i]; }
    float g = td / fmaxf(sqrtf(tw), EPS_N);
    gt[row] = fminf(fmaxf(g, -1.0f), 1.0f);
  }
}

// ---------------------------------------------------------------------------
// Kernel 3: WMMA GEMM + fused MV-AM epilogue.
// One WG = 32 weight rows (N tile) x all 512 input rows (M), K=512.
// Weight is read from HBM exactly once; output written exactly once.
// 16 waves; wave w owns output rows [32w, 32w+32) -> 2x2 WMMA 16x16 tiles.
// ---------------------------------------------------------------------------
__global__ __launch_bounds__(512) void k_gemm(const _Float16* __restrict__ a16,
                                              const float* __restrict__ weight,
                                              const long long* __restrict__ label,
                                              const float* __restrict__ gt,
                                              float* __restrict__ out) {
  constexpr int LDB = D_DIM + 8;                 // f16 elems; pad kills bank conflicts
  __shared__ _Float16 bs[TILE_N * LDB];          // ~33 KB f16 weight tile
  __shared__ float    rrec[TILE_N];              // 1/||w_row||

  const int nb      = blockIdx.x * TILE_N;
  const int t       = threadIdx.x;
  const int wave    = t >> 5;
  const int lane    = t & 31;
  const int lane_lo = lane & 15;
  const int lane_hi = lane >> 4;

  // ---- Stage weight tile fp32 -> f16 LDS, accumulate per-row sum of squares
  if (t < TILE_N) rrec[t] = 0.0f;
  __syncthreads();
  for (int r = 0; r < TILE_N; ++r) {
    float wv = weight[(size_t)(nb + r) * D_DIM + t];   // fully coalesced 2KB/row
    bs[r * LDB + t] = (_Float16)wv;
    float s = wv * wv;
#pragma unroll
    for (int off = 16; off > 0; off >>= 1) s += __shfl_xor(s, off, 32);
    if (lane == 0) atomicAdd(&rrec[r], s);             // ds_add_f32, 16 adds/row
  }
  __syncthreads();
  if (t < TILE_N) rrec[t] = 1.0f / fmaxf(sqrtf(rrec[t]), EPS_N);
  __syncthreads();

  // ---- K loop: v_wmma_f32_16x16x32_f16
  union V16 { v16h v; v8h h[2]; };
  v8f acc[2][2] = {};
  const int m0 = wave * 32;

  for (int kb = 0; kb < D_DIM; kb += 32) {
    // A fragments (16-bit A 16x32 layout: lanes<16 hold K+0..7 & K+16..23,
    // lanes>=16 hold K+8..15 & K+24..31, row M = lane_lo)
    V16 afrag[2];
#pragma unroll
    for (int i = 0; i < 2; ++i) {
      const int m = m0 + i * 16 + lane_lo;
      const _Float16* ap = a16 + (size_t)m * D_DIM + kb + lane_hi * 8;
      afrag[i].h[0] = *(const v8h*)(ap);
      afrag[i].h[1] = *(const v8h*)(ap + 16);
    }
    // B fragments from LDS: lane = column (weight row), K+8h..8h+7 & K+16+8h..16+8h+7
    V16 bfrag[2];
#pragma unroll
    for (int j = 0; j < 2; ++j) {
      const int n = j * 16 + lane_lo;
      const _Float16* bp = &bs[n * LDB + kb + lane_hi * 8];
      bfrag[j].h[0] = *(const v8h*)(bp);
      bfrag[j].h[1] = *(const v8h*)(bp + 16);
    }
#pragma unroll
    for (int i = 0; i < 2; ++i)
#pragma unroll
      for (int j = 0; j < 2; ++j)
        acc[i][j] = __builtin_amdgcn_wmma_f32_16x16x32_f16(
            false, afrag[i].v, false, bfrag[j].v,
            (short)0, acc[i][j], false, false);
  }

  // ---- Fused epilogue: normalize by ||w||, clip, MV-AM margin, scatter, scale
  // C/D layout: VGPR r -> M = r + 8*lane_hi, N = lane_lo
#pragma unroll
  for (int i = 0; i < 2; ++i) {
#pragma unroll
    for (int j = 0; j < 2; ++j) {
      const int nloc = j * 16 + lane_lo;
      const int ng   = nb + nloc;
      const float rr = rrec[nloc];
#pragma unroll
      for (int r = 0; r < 8; ++r) {
        const int m = m0 + i * 16 + lane_hi * 8 + r;
        float v = acc[i][j][r] * rr;
        v = fminf(fmaxf(v, -1.0f), 1.0f);
        const float g   = gt[m];          // L2/L1-resident, 512 floats
        const float thr = g - MARG;
        float v2 = (v > thr) ? ((TPAR + 1.0f) * v + TPAR) : v;
        if ((long long)ng == label[m]) v2 = thr;   // scatter final_gt at label col
        out[(size_t)m * C_COLS + ng] = v2 * SCALE_S;
      }
    }
  }
}

// ---------------------------------------------------------------------------
extern "C" void kernel_launch(void* const* d_in, const int* in_sizes, int n_in,
                              void* d_out, int out_size, void* d_ws, size_t ws_size,
                              hipStream_t stream) {
  const float*     input  = (const float*)d_in[0];
  const float*     weight = (const float*)d_in[1];
  const long long* label  = (const long long*)d_in[2];
  float*           out    = (float*)d_out;

  _Float16* a16 = (_Float16*)d_ws;                                   // 512 KB
  float*    gt  = (float*)((char*)d_ws + (size_t)B_ROWS * D_DIM * 2);// 2 KB

  k_norm_input<<<B_ROWS, 256, 0, stream>>>(input, a16);
  k_gt<<<B_ROWS, 256, 0, stream>>>(a16, weight, label, gt);
  k_gemm<<<C_COLS / TILE_N, 512, 0, stream>>>(a16, weight, label, gt, out);
}
